// BiLinAntisymmetricFunc_20401094656476
// MI455X (gfx1250) — matrix operand order repaired and checked by reading, not compiled, not run on hardware
//
#include <hip/hip_runtime.h>

typedef __attribute__((ext_vector_type(16))) __bf16 v16bf;
typedef __attribute__((ext_vector_type(8)))  float  v8f;
typedef __attribute__((ext_vector_type(4)))  float  f4v;

#define D_DIM 1024
#define RANK  64
#define NHEAD 8
#define NSTEP 32          // D_DIM / 32 (bf16 WMMA K=32)
#define ROWS  16          // rows of x1/x2 per block

__device__ __forceinline__ unsigned short f2bf(float f) {
  unsigned int u = __float_as_uint(f);
  unsigned int r = u + 0x7FFFu + ((u >> 16) & 1u);   // round-to-nearest-even
  return (unsigned short)(r >> 16);
}

union BFrag { uint4 q[2]; v16bf v; };

// ---------------------------------------------------------------------------
// Repack U,V (f32 [K,D,R]) into bf16 WMMA B-fragments in workspace.
// Fragment index f = ((k*NSTEP + s)*8 + m), m in 0..3 -> U col-tile m,
// m in 4..7 -> V col-tile m-4.  Each fragment = 32 lanes * 32 bytes = 1 KB.
// B layout used here (contiguous k, same as A): lane l, element e: K = 16*(l/16)+e, N = l%16.
// ---------------------------------------------------------------------------
__global__ void prep_weights_kernel(const float* __restrict__ U,
                                    const float* __restrict__ V,
                                    uint4* __restrict__ wf) {
  int gid  = blockIdx.x * 256 + threadIdx.x;  // 2048 frags * 32 lanes = 65536
  int lane = gid & 31;
  int frag = gid >> 5;
  int m = frag & 7;
  int s = (frag >> 3) & 31;
  int k = frag >> 8;
  int t = m & 3;
  const float* W = (m < 4) ? U : V;
  int hf = lane >> 4;
  int r  = t * 16 + (lane & 15);
  int dbase = s * 32 + hf * 16;
  const float* col = W + (size_t)k * D_DIM * RANK + r;
  __align__(16) unsigned short tmp[16];
#pragma unroll
  for (int e = 0; e < 16; ++e)
    tmp[e] = f2bf(col[(size_t)(dbase + e) * RANK]);
  uint4* dst = wf + ((size_t)frag * 1024 + (size_t)lane * 32) / 16;
  dst[0] = *(const uint4*)&tmp[0];
  dst[1] = *(const uint4*)&tmp[8];
}

// f32 MLP g(x): all 32 lanes cooperate, result valid in every lane.
__device__ float g_eval(const float* __restrict__ x,
                        const float* __restrict__ W1, const float* __restrict__ b1,
                        const float* __restrict__ W2, const float* __restrict__ b2,
                        const float* __restrict__ W3, const float* __restrict__ b3,
                        int lane) {
  float acc[8] = {0.f,0.f,0.f,0.f,0.f,0.f,0.f,0.f};
  for (int d = lane; d < D_DIM; d += 32) {
    float xv = x[d];
    const f4v* wr = (const f4v*)(W1 + d * 8);
    f4v w0 = wr[0], w1 = wr[1];
    acc[0] += xv * w0.x; acc[1] += xv * w0.y;
    acc[2] += xv * w0.z; acc[3] += xv * w0.w;
    acc[4] += xv * w1.x; acc[5] += xv * w1.y;
    acc[6] += xv * w1.z; acc[7] += xv * w1.w;
  }
#pragma unroll
  for (int msk = 1; msk < 32; msk <<= 1)
#pragma unroll
    for (int i = 0; i < 8; ++i) acc[i] += __shfl_xor(acc[i], msk, 32);
  float h1[8];
#pragma unroll
  for (int i = 0; i < 8; ++i) h1[i] = fmaxf(acc[i] + b1[i], 0.0f);
  float g = b3[0];
#pragma unroll
  for (int i = 0; i < 8; ++i) {
    float s = b2[i];
#pragma unroll
    for (int j = 0; j < 8; ++j) s += h1[j] * W2[j * 8 + i];
    g += fmaxf(s, 0.0f) * W3[i];
  }
  return g;
}

__global__ __launch_bounds__(256) void bilin_main_kernel(
    const float* __restrict__ x1, const float* __restrict__ x2,
    const float* __restrict__ alpha,
    const float* __restrict__ W1, const float* __restrict__ b1,
    const float* __restrict__ W2, const float* __restrict__ b2,
    const float* __restrict__ W3, const float* __restrict__ b3,
    const uint4* __restrict__ wf,
    float* __restrict__ out) {
  extern __shared__ char smem[];                 // 64 KB dynamic LDS
  unsigned short* ldsx = (unsigned short*)smem;  // [2][ROWS][D_DIM] bf16
  const int tid  = threadIdx.x;
  const int lane = tid & 31;
  const int k    = tid >> 5;                     // wave id == head id
  const long rowbase = (long)blockIdx.x * ROWS;

  // ---- stage x1/x2 tiles to LDS as bf16 (non-temporal: keep weights in L2) --
  {
    const f4v* g1p = (const f4v*)(x1 + rowbase * D_DIM);
    const f4v* g2p = (const f4v*)(x2 + rowbase * D_DIM);
    uint2* l1 = (uint2*)ldsx;
    uint2* l2 = (uint2*)(ldsx + ROWS * D_DIM);
    for (int i = tid; i < ROWS * D_DIM / 4; i += 256) {
      f4v a = __builtin_nontemporal_load(&g1p[i]);
      uint2 pa;
      pa.x = ((unsigned)f2bf(a.y) << 16) | f2bf(a.x);
      pa.y = ((unsigned)f2bf(a.w) << 16) | f2bf(a.z);
      l1[i] = pa;
      f4v b = __builtin_nontemporal_load(&g2p[i]);
      uint2 pb;
      pb.x = ((unsigned)f2bf(b.y) << 16) | f2bf(b.x);
      pb.y = ((unsigned)f2bf(b.w) << 16) | f2bf(b.z);
      l2[i] = pb;
    }
  }
  __syncthreads();

  // ---- WMMA main loop: two passes (U then V), double-buffered operands ----
  // acc[0]=x1@U, acc[1]=x2@U, acc[2]=x1@V, acc[3]=x2@V  (4 rank-tiles each)
  v8f acc[4][4];
#pragma unroll
  for (int p = 0; p < 4; ++p)
#pragma unroll
    for (int t = 0; t < 4; ++t) {
      v8f z = {0.f,0.f,0.f,0.f,0.f,0.f,0.f,0.f};
      acc[p][t] = z;
    }

  const int hf  = lane >> 4;
  const int row = lane & 15;
  const char* a1base = (const char*)ldsx + (size_t)row * (D_DIM * 2);
  const char* a2base = a1base + (size_t)ROWS * D_DIM * 2;

#pragma unroll
  for (int mat = 0; mat < 2; ++mat) {
    // fragments for this head/matrix: frag = (k*NSTEP+s)*8 + mat*4 + t
    const uint4* bmat =
        wf + ((size_t)(k * NSTEP * 8 + mat * 4) * 1024 + (size_t)lane * 32) / 16;

    BFrag Bb[2][4];
    BFrag A1b[2], A2b[2];
    // preload step 0
#pragma unroll
    for (int t = 0; t < 4; ++t) {
      Bb[0][t].q[0] = bmat[t * 64];
      Bb[0][t].q[1] = bmat[t * 64 + 1];
    }
    {
      int off = hf * 32;
      A1b[0].q[0] = *(const uint4*)(a1base + off);
      A1b[0].q[1] = *(const uint4*)(a1base + off + 16);
      A2b[0].q[0] = *(const uint4*)(a2base + off);
      A2b[0].q[1] = *(const uint4*)(a2base + off + 16);
    }

#pragma unroll 2
    for (int s = 0; s < NSTEP; ++s) {
      const int cur = s & 1, nxt = cur ^ 1;
      if (s + 1 < NSTEP) {                       // prefetch next K-step
        const uint4* bs = bmat + (size_t)(s + 1) * (8192 / 16);
#pragma unroll
        for (int t = 0; t < 4; ++t) {
          Bb[nxt][t].q[0] = bs[t * 64];
          Bb[nxt][t].q[1] = bs[t * 64 + 1];
        }
        int off = (s + 1) * 64 + hf * 32;
        A1b[nxt].q[0] = *(const uint4*)(a1base + off);
        A1b[nxt].q[1] = *(const uint4*)(a1base + off + 16);
        A2b[nxt].q[0] = *(const uint4*)(a2base + off);
        A2b[nxt].q[1] = *(const uint4*)(a2base + off + 16);
      }
#pragma unroll
      for (int t = 0; t < 4; ++t) {
        acc[mat * 2 + 0][t] = __builtin_amdgcn_wmma_f32_16x16x32_bf16(
            false, A1b[cur].v, false, Bb[cur][t].v, (short)0,
            acc[mat * 2 + 0][t], false, false);
        acc[mat * 2 + 1][t] = __builtin_amdgcn_wmma_f32_16x16x32_bf16(
            false, A2b[cur].v, false, Bb[cur][t].v, (short)0,
            acc[mat * 2 + 1][t], false, false);
      }
    }
  }
  __syncthreads();   // everyone done reading ldsx; reuse front of LDS

  float* lds_bili = (float*)smem;
  float* lds_g    = lds_bili + 16;
  if (tid < 16) lds_bili[tid] = 0.0f;
  __syncthreads();

  // ---- rank reduction: sum_r (x1U*x2V - x1V*x2U), weighted by alpha_k ----
  {
    float alpha_k = alpha[k];
    v8f p = acc[0][0] * acc[3][0] - acc[2][0] * acc[1][0];
    p    += acc[0][1] * acc[3][1] - acc[2][1] * acc[1][1];
    p    += acc[0][2] * acc[3][2] - acc[2][2] * acc[1][2];
    p    += acc[0][3] * acc[3][3] - acc[2][3] * acc[1][3];
#pragma unroll
    for (int j = 0; j < 8; ++j) {
      float v = p[j];
#pragma unroll
      for (int msk = 1; msk < 16; msk <<= 1) v += __shfl_xor(v, msk, 32);
      if ((lane & 15) == 0) {
        int r = j + (lane >> 4) * 8;       // C-layout: VGPR j = rows j, j+8
        atomicAdd(&lds_bili[r], alpha_k * v);
      }
    }
  }

  // ---- f32 MLP g(x1)-g(x2): wave k handles rows 2k, 2k+1 ----
  for (int rr = 0; rr < 2; ++rr) {
    int r = 2 * k + rr;
    float g1 = g_eval(x1 + (rowbase + r) * D_DIM, W1, b1, W2, b2, W3, b3, lane);
    float g2 = g_eval(x2 + (rowbase + r) * D_DIM, W1, b1, W2, b2, W3, b3, lane);
    if (lane == 0) lds_g[r] = g1 - g2;
  }
  __syncthreads();

  if (tid < 16) out[rowbase + tid] = lds_g[tid] + lds_bili[tid];
}

extern "C" void kernel_launch(void* const* d_in, const int* in_sizes, int n_in,
                              void* d_out, int out_size, void* d_ws, size_t ws_size,
                              hipStream_t stream) {
  const float* x1    = (const float*)d_in[0];
  const float* x2    = (const float*)d_in[1];
  const float* U     = (const float*)d_in[2];
  const float* V     = (const float*)d_in[3];
  const float* alpha = (const float*)d_in[4];
  const float* W1    = (const float*)d_in[5];
  const float* b1    = (const float*)d_in[6];
  const float* W2    = (const float*)d_in[7];
  const float* b2    = (const float*)d_in[8];
  const float* W3    = (const float*)d_in[9];
  const float* b3    = (const float*)d_in[10];
  float* out = (float*)d_out;
  uint4* wf  = (uint4*)d_ws;       // 2 MB of repacked bf16 weight fragments

  prep_weights_kernel<<<256, 256, 0, stream>>>(U, V, wf);

  int nblocks = out_size / ROWS;   // 65536 / 16 = 4096
  bilin_main_kernel<<<nblocks, 256, 64 * 1024, stream>>>(
      x1, x2, alpha, W1, b1, W2, b2, W3, b3, wf, out);
}
